// MalaAttention_19481971654868
// MI455X (gfx1250) — matrix operand order repaired and checked by reading, not compiled, not run on hardware
//
#include <hip/hip_runtime.h>
#include <hip/hip_bf16.h>

// ---------------------------------------------------------------------------
// MALA attention for MI455X (gfx1250, wave32, WMMA).
// Pipeline:
//   1. f32->bf16 converts (hidden_states, 5 weight matrices)
//   2. 4x projection GEMMs (bf16 WMMA, f32 accum): q,k,v,gate
//   3. rope_pack: RoPE on q/k, write q_bf16 [B,S,H,DK], kT/vT bf16 [B,H,D,S]
//   4. row sums (ksum/vsum), den = scale*(q.ksum)+S
//   5. kv GEMM per (b,h): kvT[e,d] = sum_s v[s,e]k[s,d]  (WMMA, K=4096)
//   6. num GEMM per (b,h) fused epilogue: +vsum, /den, +LePE conv5, *gate -> bf16
//   7. output projection GEMM -> d_out (f32)
// GEMM tile staging uses GLOBAL_LOAD_ASYNC_TO_LDS_B128 (ASYNCcnt) when the
// toolchain exposes the builtins; otherwise falls back to ld+ds_store.
// ---------------------------------------------------------------------------

typedef __attribute__((ext_vector_type(16))) __bf16 v16bf;
typedef __attribute__((ext_vector_type(8)))  float  v8f;
typedef __attribute__((ext_vector_type(4)))  int    v4i;

#define B_  4
#define S_  4096
#define D_  1024
#define H_  8
#define DK_ 128
#define DV_ 128

#if __has_builtin(__builtin_amdgcn_global_load_async_to_lds_b128) && \
    __has_builtin(__builtin_amdgcn_s_wait_asynccnt)
#define HAVE_ASYNC_LDS 1
#else
#define HAVE_ASYNC_LDS 0
#endif

__device__ __forceinline__ __bf16 f2bf(float f) {
  union { float f; unsigned u; } x; x.f = f;
  unsigned r = x.u + 0x7FFFu + ((x.u >> 16) & 1u);   // round-to-nearest-even
  unsigned short h = (unsigned short)(r >> 16);
  return __builtin_bit_cast(__bf16, h);
}
__device__ __forceinline__ float bf2f(__bf16 b) {
  unsigned u = ((unsigned)__builtin_bit_cast(unsigned short, b)) << 16;
  return __builtin_bit_cast(float, u);
}

#if HAVE_ASYNC_LDS
// 16B global -> LDS, no VGPR round trip; tracked by ASYNCcnt.
// ROCm builtin signature (from probe): (global int4*, shared int4*, imm, imm)
__device__ __forceinline__ void cp_async16(const __bf16* g, __bf16* l) {
  __builtin_amdgcn_global_load_async_to_lds_b128(
      (__attribute__((address_space(1))) v4i*)g,
      (__attribute__((address_space(3))) v4i*)l, 0, 0);
}
#endif

struct U4x2 { uint4 x, y; };

// A fragment (16x32 bf16): lane<16 -> K{0..7,16..23}, lane>=16 -> K{8..15,24..31}
__device__ __forceinline__ v16bf load_fragA(const __bf16* rowbase, int sel) {
  const uint4* p = (const uint4*)rowbase;          // row = 32 bf16 = 4x uint4
  U4x2 t; t.x = p[sel]; t.y = p[2 + sel];
  return __builtin_bit_cast(v16bf, t);
}
// B fragment (32x16 bf16): lane<16 -> K{0..7,16..23}, lane>=16 -> K{8..15,24..31} (N = lane&15)
__device__ __forceinline__ v16bf load_fragB(const __bf16* rowbase, int sel) {
  const uint4* p = (const uint4*)rowbase;
  U4x2 t; t.x = p[sel]; t.y = p[2 + sel];
  return __builtin_bit_cast(v16bf, t);
}

// Shared 128x128-tile double-buffered WMMA mainloop.
// A: [rows, K] (row stride lda), operand rows = output rows (row0..row0+127)
// Bw:[cols, K] (row stride ldb), operand rows = output cols (col0..col0+127)
// 256 threads = 8 waves; wave (wid&3, wid>>2) owns a 32x64 patch = 2x4 tiles.
__device__ __forceinline__ void wmma_mainloop_128x128(
    const __bf16* __restrict__ A,  long lda,
    const __bf16* __restrict__ Bw, long ldb,
    int K, int row0, int col0,
    __bf16 (&As)[2][128 * 32], __bf16 (&Bs)[2][128 * 32],
    v8f (&acc)[2][4])
{
  const int tid = threadIdx.x, lane = tid & 31, wid = tid >> 5;
  const int wm = (wid & 3) * 32, wn = (wid >> 2) * 64;
  const int laneRow = lane & 15, sel = lane >> 4;

  // Staging: 512 16B chunks per tile over 256 threads -> 2 chunks/thread.
  const int id0 = tid * 2, id1 = tid * 2 + 1;
  const int r0 = id0 >> 2, c0 = (id0 & 3) * 8;
  const int r1 = id1 >> 2, c1 = (id1 & 3) * 8;

  const int nk = K >> 5;

#if HAVE_ASYNC_LDS
  cp_async16(A  + (size_t)(row0 + r0) * lda + c0, &As[0][r0 * 32 + c0]);
  cp_async16(A  + (size_t)(row0 + r1) * lda + c1, &As[0][r1 * 32 + c1]);
  cp_async16(Bw + (size_t)(col0 + r0) * ldb + c0, &Bs[0][r0 * 32 + c0]);
  cp_async16(Bw + (size_t)(col0 + r1) * ldb + c1, &Bs[0][r1 * 32 + c1]);
  __builtin_amdgcn_s_wait_asynccnt(0);
  __syncthreads();
#else
  uint4 a0 = *(const uint4*)(A  + (size_t)(row0 + r0) * lda + c0);
  uint4 a1 = *(const uint4*)(A  + (size_t)(row0 + r1) * lda + c1);
  uint4 b0 = *(const uint4*)(Bw + (size_t)(col0 + r0) * ldb + c0);
  uint4 b1 = *(const uint4*)(Bw + (size_t)(col0 + r1) * ldb + c1);
  *(uint4*)&As[0][r0 * 32 + c0] = a0; *(uint4*)&As[0][r1 * 32 + c1] = a1;
  *(uint4*)&Bs[0][r0 * 32 + c0] = b0; *(uint4*)&Bs[0][r1 * 32 + c1] = b1;
  __syncthreads();
#endif

  for (int kk = 0; kk < nk; ++kk) {
    const int cur = kk & 1, nxt = cur ^ 1;

#if HAVE_ASYNC_LDS
    if (kk + 1 < nk) {   // kick next tile straight into LDS[nxt]
      const int k0 = (kk + 1) << 5;
      cp_async16(A  + (size_t)(row0 + r0) * lda + k0 + c0, &As[nxt][r0 * 32 + c0]);
      cp_async16(A  + (size_t)(row0 + r1) * lda + k0 + c1, &As[nxt][r1 * 32 + c1]);
      cp_async16(Bw + (size_t)(col0 + r0) * ldb + k0 + c0, &Bs[nxt][r0 * 32 + c0]);
      cp_async16(Bw + (size_t)(col0 + r1) * ldb + k0 + c1, &Bs[nxt][r1 * 32 + c1]);
    }
#else
    uint4 na0, na1, nb0, nb1;
    if (kk + 1 < nk) {
      const int k0 = (kk + 1) << 5;
      na0 = *(const uint4*)(A  + (size_t)(row0 + r0) * lda + k0 + c0);
      na1 = *(const uint4*)(A  + (size_t)(row0 + r1) * lda + k0 + c1);
      nb0 = *(const uint4*)(Bw + (size_t)(col0 + r0) * ldb + k0 + c0);
      nb1 = *(const uint4*)(Bw + (size_t)(col0 + r1) * ldb + k0 + c1);
    }
#endif
    if (kk + 2 < nk) {   // gfx1250 global_prefetch_b8 for the tile after next
      const int k2 = (kk + 2) << 5;
      __builtin_prefetch(A  + (size_t)(row0 + r0) * lda + k2 + c0, 0, 1);
      __builtin_prefetch(Bw + (size_t)(col0 + r0) * ldb + k2 + c0, 0, 1);
    }

    v16bf af[2], bfv[4];
#pragma unroll
    for (int i = 0; i < 2; ++i)
      af[i] = load_fragA(&As[cur][(wm + i * 16 + laneRow) * 32], sel);
#pragma unroll
    for (int j = 0; j < 4; ++j)
      bfv[j] = load_fragB(&Bs[cur][(wn + j * 16 + laneRow) * 32], sel);
#pragma unroll
    for (int i = 0; i < 2; ++i)
#pragma unroll
      for (int j = 0; j < 4; ++j)
        acc[i][j] = __builtin_amdgcn_wmma_f32_16x16x32_bf16(
            false, af[i], false, bfv[j], (short)0, acc[i][j], false, false);

#if HAVE_ASYNC_LDS
    if (kk + 1 < nk) __builtin_amdgcn_s_wait_asynccnt(0);
#else
    if (kk + 1 < nk) {
      *(uint4*)&As[nxt][r0 * 32 + c0] = na0; *(uint4*)&As[nxt][r1 * 32 + c1] = na1;
      *(uint4*)&Bs[nxt][r0 * 32 + c0] = nb0; *(uint4*)&Bs[nxt][r1 * 32 + c1] = nb1;
    }
#endif
    __syncthreads();
  }
}

// ---------------------------------------------------------------------------
// Kernels
// ---------------------------------------------------------------------------

__global__ void f32_to_bf16_kernel(const float* __restrict__ x,
                                   __bf16* __restrict__ y, long n) {
  for (long i = (long)blockIdx.x * blockDim.x + threadIdx.x; i < n;
       i += (long)gridDim.x * blockDim.x)
    y[i] = f2bf(x[i]);
}

// C[M,N] (f32) = A[M,K](bf16) * W[N,K]^T(bf16)
__global__ __launch_bounds__(256) void gemm_proj_kernel(
    const __bf16* __restrict__ A, const __bf16* __restrict__ W,
    float* __restrict__ C, int M, int N, int K) {
  __shared__ __bf16 As[2][128 * 32];
  __shared__ __bf16 Bs[2][128 * 32];
  const int tilesN = N >> 7;
  const int row0 = (blockIdx.x / tilesN) * 128;
  const int col0 = (blockIdx.x % tilesN) * 128;

  v8f z = {0.f, 0.f, 0.f, 0.f, 0.f, 0.f, 0.f, 0.f};
  v8f acc[2][4] = {{z, z, z, z}, {z, z, z, z}};
  wmma_mainloop_128x128(A, K, W, K, K, row0, col0, As, Bs, acc);

  const int lane = threadIdx.x & 31, wid = threadIdx.x >> 5;
  const int wm = (wid & 3) * 32, wn = (wid >> 2) * 64;
  const int sel = lane >> 4, ln = lane & 15;
#pragma unroll
  for (int i = 0; i < 2; ++i)
#pragma unroll
    for (int j = 0; j < 4; ++j) {
      union { v8f v; float f[8]; } u; u.v = acc[i][j];
#pragma unroll
      for (int r = 0; r < 8; ++r) {
        const int m = row0 + wm + i * 16 + sel * 8 + r;
        const int n = col0 + wn + j * 16 + ln;
        C[(size_t)m * N + n] = u.f[r];
      }
    }
}

// RoPE on q,k; emit q bf16 [B,S,H,DK], kT/vT bf16 [B,H,D,S]
__global__ void rope_pack_kernel(const float* __restrict__ q32,
                                 const float* __restrict__ k32,
                                 const float* __restrict__ v32,
                                 __bf16* __restrict__ qb,
                                 __bf16* __restrict__ kT,
                                 __bf16* __restrict__ vT) {
  const long idx = (long)blockIdx.x * blockDim.x + threadIdx.x; // (b,s,h,d)
  const int d = idx & 127;
  const int h = (idx >> 7) & 7;
  const int s = (idx >> 10) & 4095;
  const int b = (int)(idx >> 22);
  const int j = d & 63;
  // inv_freq[j] = 10000^(-j/64) -> angle = s * exp(-j * ln(1e4)/64)
  const float ang = (float)s * __expf(-(float)j * 0.14391156f);
  const float sn = __sinf(ang), cs = __cosf(ang);
  const float qv = q32[idx], kv = k32[idx], vv = v32[idx];
  float qrot, krot;
  if (d < 64) { qrot = -q32[idx + 64]; krot = -k32[idx + 64]; }
  else        { qrot =  q32[idx - 64]; krot =  k32[idx - 64]; }
  const float qo = qv * cs + qrot * sn;
  const float ko = kv * cs + krot * sn;
  qb[idx] = f2bf(qo);
  const size_t t = ((size_t)(b * H_ + h) * DK_ + d) * S_ + s;
  kT[t] = f2bf(ko);
  vT[t] = f2bf(vv);
}

// one wave per (b,h,d) row: ksum/vsum over S
__global__ void row_sums_kernel(const __bf16* __restrict__ kT,
                                const __bf16* __restrict__ vT,
                                float* __restrict__ ksum,
                                float* __restrict__ vsum) {
  const int row = blockIdx.x, lane = threadIdx.x;
  const size_t base = (size_t)row * S_;
  float a = 0.f, c = 0.f;
  for (int i = lane; i < S_; i += 32) { a += bf2f(kT[base + i]); c += bf2f(vT[base + i]); }
#pragma unroll
  for (int off = 16; off > 0; off >>= 1) {
    a += __shfl_down(a, off, 32);
    c += __shfl_down(c, off, 32);
  }
  if (lane == 0) { ksum[row] = a; vsum[row] = c; }
}

// one wave per (b,s,h): den = scale * q.ksum + S
__global__ void den_kernel(const __bf16* __restrict__ qb,
                           const float* __restrict__ ksum,
                           float* __restrict__ den, float scale) {
  const int r = blockIdx.x;                 // (b*S+s)*H + h
  const int h = r & 7;
  const int b = r >> 15;                    // r / (S_*H_)
  const int lane = threadIdx.x;
  const __bf16* q = qb + (size_t)r * DK_;
  const float* ks = ksum + ((size_t)b * H_ + h) * DK_;
  float a = 0.f;
#pragma unroll
  for (int i = lane; i < DK_; i += 32) a += bf2f(q[i]) * ks[i];
#pragma unroll
  for (int off = 16; off > 0; off >>= 1) a += __shfl_down(a, off, 32);
  if (lane == 0) den[r] = a * scale + (float)S_;
}

// kvT[b,h][e,d] = sum_s v[s,e] * k[s,d]   (A=vT row e, B=kT row d, K=S)
__global__ __launch_bounds__(256) void kv_gemm_kernel(
    const __bf16* __restrict__ vT, const __bf16* __restrict__ kT,
    __bf16* __restrict__ kvT) {
  __shared__ __bf16 As[2][128 * 32];
  __shared__ __bf16 Bs[2][128 * 32];
  const int bh = blockIdx.x;
  const __bf16* A  = vT + (size_t)bh * DV_ * S_;
  const __bf16* Bw = kT + (size_t)bh * DK_ * S_;
  v8f z = {0.f, 0.f, 0.f, 0.f, 0.f, 0.f, 0.f, 0.f};
  v8f acc[2][4] = {{z, z, z, z}, {z, z, z, z}};
  wmma_mainloop_128x128(A, S_, Bw, S_, S_, 0, 0, As, Bs, acc);

  const int lane = threadIdx.x & 31, wid = threadIdx.x >> 5;
  const int wm = (wid & 3) * 32, wn = (wid >> 2) * 64;
  const int sel = lane >> 4, ln = lane & 15;
  __bf16* out = kvT + (size_t)bh * DV_ * DK_;
#pragma unroll
  for (int i = 0; i < 2; ++i)
#pragma unroll
    for (int j = 0; j < 4; ++j) {
      union { v8f v; float f[8]; } u; u.v = acc[i][j];
#pragma unroll
      for (int r = 0; r < 8; ++r) {
        const int e = wm + i * 16 + sel * 8 + r;
        const int d = wn + j * 16 + ln;
        out[(size_t)e * DK_ + d] = f2bf(u.f[r]);
      }
    }
}

// num = q @ kv with fused: *scale, +vsum, /den, +LePE conv5, *gate -> bf16
__global__ __launch_bounds__(256) void num_gemm_fused_kernel(
    const __bf16* __restrict__ qb, const __bf16* __restrict__ kvT,
    const float* __restrict__ vsum, const float* __restrict__ den,
    const float* __restrict__ v32, const float* __restrict__ g32,
    const float* __restrict__ lepe_w, const float* __restrict__ lepe_b,
    __bf16* __restrict__ ob, float scale) {
  __shared__ __bf16 As[2][128 * 32];
  __shared__ __bf16 Bs[2][128 * 32];
  const int mt = blockIdx.x & 31;          // S tile (S/128 = 32)
  const int bh = blockIdx.x >> 5;
  const int b = bh >> 3, h = bh & 7;

  // q rows: s stride = H*DK = 1024 ; kv operand row e over d (ldb = 128)
  const __bf16* A  = qb + ((size_t)b * S_ * H_ + h) * DK_;
  const __bf16* Bw = kvT + (size_t)bh * DV_ * DK_;
  v8f z = {0.f, 0.f, 0.f, 0.f, 0.f, 0.f, 0.f, 0.f};
  v8f acc[2][4] = {{z, z, z, z}, {z, z, z, z}};
  wmma_mainloop_128x128(A, H_ * DK_, Bw, DK_, DK_, mt * 128, 0, As, Bs, acc);

  const int lane = threadIdx.x & 31, wid = threadIdx.x >> 5;
  const int wm = (wid & 3) * 32, wn = (wid >> 2) * 64;
  const int sel = lane >> 4, ln = lane & 15;
#pragma unroll
  for (int i = 0; i < 2; ++i)
#pragma unroll
    for (int j = 0; j < 4; ++j) {
      union { v8f v; float f[8]; } u; u.v = acc[i][j];
#pragma unroll
      for (int r = 0; r < 8; ++r) {
        const int s = mt * 128 + wm + i * 16 + sel * 8 + r;
        const int e = wn + j * 16 + ln;
        const int c = h * DV_ + e;
        const float numv = u.f[r] * scale + vsum[bh * DV_ + e];
        const float attn = numv / den[((size_t)b * S_ + s) * H_ + h];
        float lep = lepe_b[c];
#pragma unroll
        for (int t = 0; t < 5; ++t) {
          const int ss = s + t - 2;
          if (ss >= 0 && ss < S_)
            lep += v32[((size_t)(b * S_ + ss) * H_ + h) * DV_ + e] * lepe_w[c * 5 + t];
        }
        const float o = (attn + lep) * g32[((size_t)b * S_ + s) * D_ + c];
        ob[((size_t)b * S_ + s) * D_ + c] = f2bf(o);
      }
    }
}

// ---------------------------------------------------------------------------
// Launch
// ---------------------------------------------------------------------------
extern "C" void kernel_launch(void* const* d_in, const int* in_sizes, int n_in,
                              void* d_out, int out_size, void* d_ws, size_t ws_size,
                              hipStream_t stream) {
  (void)in_sizes; (void)n_in; (void)out_size; (void)ws_size;
  const float* hs  = (const float*)d_in[0];
  const float* Wq  = (const float*)d_in[1];
  const float* Wk  = (const float*)d_in[2];
  const float* Wv  = (const float*)d_in[3];
  const float* Wg  = (const float*)d_in[4];
  const float* Wo  = (const float*)d_in[5];
  const float* lw  = (const float*)d_in[6];
  const float* lb  = (const float*)d_in[7];
  float* out = (float*)d_out;

  const long BSD = (long)B_ * S_ * D_;          // 16,777,216
  const long WN  = (long)D_ * D_;               // 1,048,576

  char* w = (char*)d_ws;
  size_t off = 0;
  auto take = [&](size_t bytes) { char* p = w + off; off = (off + bytes + 255) & ~(size_t)255; return p; };

  __bf16* hsb  = (__bf16*)take(BSD * 2);
  __bf16* Wb[5];
  for (int i = 0; i < 5; ++i) Wb[i] = (__bf16*)take(WN * 2);
  float* q32 = (float*)take(BSD * 4);
  float* k32 = (float*)take(BSD * 4);
  float* v32 = (float*)take(BSD * 4);
  float* g32 = (float*)take(BSD * 4);
  __bf16* qb  = (__bf16*)take(BSD * 2);
  __bf16* kTb = (__bf16*)take(BSD * 2);
  __bf16* vTb = (__bf16*)take(BSD * 2);
  __bf16* kvT = (__bf16*)take((size_t)B_ * H_ * DV_ * DK_ * 2);
  float* ksum = (float*)take((size_t)B_ * H_ * DK_ * 4);
  float* vsum = (float*)take((size_t)B_ * H_ * DV_ * 4);
  float* den  = (float*)take((size_t)B_ * S_ * H_ * 4);
  __bf16* ob  = (__bf16*)take(BSD * 2);

  const float scale = 0.08838834764831845f;     // DK^-0.5

  // 1) converts
  f32_to_bf16_kernel<<<8192, 256, 0, stream>>>(hs, hsb, BSD);
  const float* Wsrc[5] = {Wq, Wk, Wv, Wg, Wo};
  for (int i = 0; i < 5; ++i)
    f32_to_bf16_kernel<<<2048, 256, 0, stream>>>(Wsrc[i], Wb[i], WN);

  // 2) projections (M=16384, N=1024, K=1024) -> 128x8 = 1024 blocks
  gemm_proj_kernel<<<1024, 256, 0, stream>>>(hsb, Wb[0], q32, B_ * S_, D_, D_);
  gemm_proj_kernel<<<1024, 256, 0, stream>>>(hsb, Wb[1], k32, B_ * S_, D_, D_);
  gemm_proj_kernel<<<1024, 256, 0, stream>>>(hsb, Wb[2], v32, B_ * S_, D_, D_);
  gemm_proj_kernel<<<1024, 256, 0, stream>>>(hsb, Wb[3], g32, B_ * S_, D_, D_);

  // 3) RoPE + transposed bf16 packs
  rope_pack_kernel<<<BSD / 256, 256, 0, stream>>>(q32, k32, v32, qb, kTb, vTb);

  // 4) reductions
  row_sums_kernel<<<B_ * H_ * DK_, 32, 0, stream>>>(kTb, vTb, ksum, vsum);
  den_kernel<<<B_ * S_ * H_, 32, 0, stream>>>(qb, ksum, den, scale);

  // 5) kv = K^T V per (b,h)
  kv_gemm_kernel<<<B_ * H_, 256, 0, stream>>>(vTb, kTb, kvT);

  // 6) num GEMM + fused normalize/LePE/gate
  num_gemm_fused_kernel<<<B_ * H_ * (S_ / 128), 256, 0, stream>>>(
      qb, kvT, vsum, den, v32, g32, lw, lb, ob, scale);

  // 7) output projection -> f32 out
  gemm_proj_kernel<<<1024, 256, 0, stream>>>(ob, Wb[4], out, B_ * S_, D_, D_);
}
